// TransducerJointNetworkV1_49512382988585
// MI455X (gfx1250) — compile-verified, matrix operand order checked by reading, not run on hardware
//
#include <hip/hip_runtime.h>

// ---------------------------------------------------------------------------
// Problem constants: B=4, T=256, S=128, E=512, P=640, H=512, V=1024
// ---------------------------------------------------------------------------
#define Bdim 4
#define Tdim 256
#define Sdim 128
#define Edim 512
#define Pdim 640
#define Hdim 512
#define Vdim 1024
#define MTILE 64            // s-rows per workgroup (4 WMMA M-subtiles)
#define NWAVES 16           // 512 threads
#define NMI 4               // M-subtiles per wave (all of them)
#define NNJ 4               // N-subtiles per wave (64 columns)

typedef __attribute__((ext_vector_type(16))) __bf16 v16bf;
typedef __attribute__((ext_vector_type(8)))  float  v8f;

struct Frag256 { unsigned int w[8]; };   // 32 bytes == v16bf

__device__ __forceinline__ unsigned short f2bf(float f) {
    unsigned int u = __float_as_uint(f);
    unsigned int r = u + 0x7FFFu + ((u >> 16) & 1u);   // round-to-nearest-even
    return (unsigned short)(r >> 16);
}

// ---------------------------------------------------------------------------
// Small fp32 projection kernel: Y[r,h] = X[r,:]·W[h,:] + bias[h]
// ---------------------------------------------------------------------------
__global__ __launch_bounds__(256)
void proj_kernel(const float* __restrict__ X, const float* __restrict__ W,
                 const float* __restrict__ bias, float* __restrict__ Y,
                 int rows, int K, int H) {
    int idx = blockIdx.x * 256 + threadIdx.x;
    if (idx >= rows * H) return;
    int r = idx / H;
    int h = idx - r * H;
    const float4* x4 = (const float4*)(X + (size_t)r * K);
    const float4* w4 = (const float4*)(W + (size_t)h * K);
    float acc = bias ? bias[h] : 0.0f;
    int k4 = K >> 2;
    for (int k = 0; k < k4; ++k) {
        float4 a = x4[k];
        float4 b = w4[k];
        acc = fmaf(a.x, b.x, acc);
        acc = fmaf(a.y, b.y, acc);
        acc = fmaf(a.z, b.z, acc);
        acc = fmaf(a.w, b.w, acc);
    }
    Y[idx] = acc;
}

// ---------------------------------------------------------------------------
// fp32 -> bf16 convert (W2 staged once, 1 MB bf16 -> L2 resident)
// ---------------------------------------------------------------------------
__global__ __launch_bounds__(256)
void cvt_bf16_kernel(const float* __restrict__ X, unsigned short* __restrict__ Y, int n) {
    int i = blockIdx.x * 256 + threadIdx.x;
    if (i < n) Y[i] = f2bf(X[i]);
}

// ---------------------------------------------------------------------------
// Fused joint kernel. One workgroup = (b, t, 64-row s tile) x full V=1024.
// 16 waves; wave w owns 64 columns (4 N-subtiles) x ALL 4 M-subtiles
// -> 16 accumulators/wave; each B fragment feeds 4 WMMAs from registers.
// A tile (64 KB) stored in LDS pre-permuted into WMMA A-fragment order
// (conflict-free ds_load_b128); softmax scratch aliases the dead A region.
// ---------------------------------------------------------------------------
__global__ __launch_bounds__(512)
void joint_kernel(const float* __restrict__ encP,          // [B*T, H]
                  const float* __restrict__ predP,         // [B*S, H]
                  const unsigned short* __restrict__ W2bf, // [V, H] bf16
                  const float* __restrict__ b2,            // [V]
                  float* __restrict__ out)                 // [B*T*S, V]
{
    // 64 KB shared: A tile (fragment order) during GEMM, reduction buf after.
    __shared__ __align__(16) char shraw[NMI * 16 * Hdim * 2];
    unsigned short* tileA = (unsigned short*)shraw;        // [mi][ks][lane][16]
    float* redbuf = (float*)shraw;                         // [64 rows][16 waves]
    __shared__ float gmax_sh[MTILE];
    __shared__ float glse_sh[MTILE];

    const int tid  = threadIdx.x;
    const int bid  = blockIdx.x;
    const int sblocks = Sdim / MTILE;            // 2
    const int sblk = bid % sblocks;
    const int bt   = bid / sblocks;              // b*T + t
    const int b    = bt / Tdim;
    const int s0   = sblk * MTILE;

    const float* enc  = encP  + (size_t)bt * Hdim;
    const float* pred = predP + (size_t)(b * Sdim + s0) * Hdim;

    // ---- Phase 1: combined = tanh(enc + pred), bf16, permuted into LDS ----
    // A-frag layout: lane = rr + 16*laneHi ; halves 0..7 = K {laneHi*8..+7},
    // halves 8..15 = K {16+laneHi*8..+7} within each 32-wide k-step.
    for (int i = tid; i < MTILE * Hdim; i += 512) {
        int r = i >> 9;              // 0..63
        int c = i & (Hdim - 1);
        float v = tanhf(enc[c] + pred[(size_t)r * Hdim + c]);
        int mi     = r >> 4;
        int rr     = r & 15;
        int ks     = c >> 5;
        int within = c & 31;
        int laneHi = (within >> 3) & 1;
        int h      = (within & 7) + ((within & 16) >> 1);
        tileA[mi * 8192 + ks * 512 + (rr + (laneHi << 4)) * 16 + h] = f2bf(v);
    }
    __syncthreads();

    const int wave     = tid >> 5;
    const int lane     = tid & 31;
    const int laneRow  = lane & 15;
    const int laneHi   = lane >> 4;
    const int ncolBase = wave * 64;

    v8f acc[NMI][NNJ] = {};   // [M-subtile][N-subtile]

    // ---- Phase 2: K loop (16 steps of K=32); 16 WMMA per step ----
    for (int ks = 0; ks < 16; ++ks) {
        v16bf aV[NMI];
        #pragma unroll
        for (int mi = 0; mi < NMI; ++mi) {
            const unsigned short* ap = tileA + mi * 8192 + ks * 512 + lane * 16;
            uint4 alo = *(const uint4*)(ap);       // halves 0..7
            uint4 ahi = *(const uint4*)(ap + 8);   // halves 8..15
            Frag256 fa = { { alo.x, alo.y, alo.z, alo.w, ahi.x, ahi.y, ahi.z, ahi.w } };
            aV[mi] = __builtin_bit_cast(v16bf, fa);
        }
        #pragma unroll
        for (int j = 0; j < NNJ; ++j) {
            // B fragment: B[k,n] = W2[n,k]; n on lanes, K contiguous per lane
            const unsigned short* bp =
                W2bf + (size_t)(ncolBase + j * 16 + laneRow) * Hdim + ks * 32 + laneHi * 8;
            uint4 blo = *(const uint4*)(bp);
            uint4 bhi = *(const uint4*)(bp + 16);
            Frag256 fb = { { blo.x, blo.y, blo.z, blo.w, bhi.x, bhi.y, bhi.z, bhi.w } };
            v16bf bV = __builtin_bit_cast(v16bf, fb);
            #pragma unroll
            for (int mi = 0; mi < NMI; ++mi) {
                acc[mi][j] = __builtin_amdgcn_wmma_f32_16x16x32_bf16(
                    false, aV[mi], false, bV, (short)0, acc[mi][j], false, false);
            }
        }
    }

    // ---- Phase 3: bias add ----
    #pragma unroll
    for (int j = 0; j < NNJ; ++j) {
        float bb = b2[ncolBase + j * 16 + laneRow];
        #pragma unroll
        for (int mi = 0; mi < NMI; ++mi)
            #pragma unroll
            for (int r = 0; r < 8; ++r) acc[mi][j][r] += bb;
    }

    __syncthreads();   // A tile dead; redbuf aliases it from here on

    // C/D layout: acc[mi][j][r] = row (mi*16 + laneHi*8 + r), col (j*16 + laneRow)
    // ---- Phase 4a: row max (lane-local -> shfl over 16-lane group ->
    //      per-wave partial in LDS -> 64 threads reduce one row each) ----
    float rmax[NMI][8];
    #pragma unroll
    for (int mi = 0; mi < NMI; ++mi)
        #pragma unroll
        for (int r = 0; r < 8; ++r) {
            float m = acc[mi][0][r];
            #pragma unroll
            for (int j = 1; j < NNJ; ++j) m = fmaxf(m, acc[mi][j][r]);
            rmax[mi][r] = m;
        }
    #pragma unroll
    for (int mask = 1; mask < 16; mask <<= 1)
        #pragma unroll
        for (int mi = 0; mi < NMI; ++mi)
            #pragma unroll
            for (int r = 0; r < 8; ++r)
                rmax[mi][r] = fmaxf(rmax[mi][r], __shfl_xor(rmax[mi][r], mask, 32));
    if (laneRow == 0) {
        #pragma unroll
        for (int mi = 0; mi < NMI; ++mi)
            #pragma unroll
            for (int r = 0; r < 8; ++r)
                redbuf[(mi * 16 + laneHi * 8 + r) * NWAVES + wave] = rmax[mi][r];
    }
    __syncthreads();
    if (tid < MTILE) {
        float m = redbuf[tid * NWAVES];
        #pragma unroll
        for (int w = 1; w < NWAVES; ++w) m = fmaxf(m, redbuf[tid * NWAVES + w]);
        gmax_sh[tid] = m;
    }
    __syncthreads();

    // ---- Phase 4b: sum(exp(x - gmax)) ----
    float gmax[NMI][8];
    float rsum[NMI][8];
    #pragma unroll
    for (int mi = 0; mi < NMI; ++mi)
        #pragma unroll
        for (int r = 0; r < 8; ++r) {
            float gm = gmax_sh[mi * 16 + laneHi * 8 + r];
            gmax[mi][r] = gm;
            float s = 0.0f;
            #pragma unroll
            for (int j = 0; j < NNJ; ++j) s += __expf(acc[mi][j][r] - gm);
            rsum[mi][r] = s;
        }
    #pragma unroll
    for (int mask = 1; mask < 16; mask <<= 1)
        #pragma unroll
        for (int mi = 0; mi < NMI; ++mi)
            #pragma unroll
            for (int r = 0; r < 8; ++r)
                rsum[mi][r] += __shfl_xor(rsum[mi][r], mask, 32);
    __syncthreads();   // redbuf reuse (max partials consumed)
    if (laneRow == 0) {
        #pragma unroll
        for (int mi = 0; mi < NMI; ++mi)
            #pragma unroll
            for (int r = 0; r < 8; ++r)
                redbuf[(mi * 16 + laneHi * 8 + r) * NWAVES + wave] = rsum[mi][r];
    }
    __syncthreads();
    if (tid < MTILE) {
        float s = redbuf[tid * NWAVES];
        #pragma unroll
        for (int w = 1; w < NWAVES; ++w) s += redbuf[tid * NWAVES + w];
        glse_sh[tid] = gmax_sh[tid] + __logf(s);
    }
    __syncthreads();

    // ---- Phase 5: single HBM write of log_softmax ----
    float glse[NMI][8];
    #pragma unroll
    for (int mi = 0; mi < NMI; ++mi)
        #pragma unroll
        for (int r = 0; r < 8; ++r)
            glse[mi][r] = glse_sh[mi * 16 + laneHi * 8 + r];

    size_t outBase = ((size_t)bt * Sdim + s0) * Vdim;
    #pragma unroll
    for (int mi = 0; mi < NMI; ++mi)
        #pragma unroll
        for (int j = 0; j < NNJ; ++j) {
            int ncol = ncolBase + j * 16 + laneRow;
            #pragma unroll
            for (int r = 0; r < 8; ++r) {
                int row = mi * 16 + laneHi * 8 + r;
                out[outBase + (size_t)row * Vdim + ncol] = acc[mi][j][r] - glse[mi][r];
            }
        }
}

// ---------------------------------------------------------------------------
// Launch
// ---------------------------------------------------------------------------
extern "C" void kernel_launch(void* const* d_in, const int* in_sizes, int n_in,
                              void* d_out, int out_size, void* d_ws, size_t ws_size,
                              hipStream_t stream) {
    const float* src = (const float*)d_in[0];  // [B,T,E]
    const float* tgt = (const float*)d_in[1];  // [B,S,P]
    const float* We  = (const float*)d_in[2];  // [H,E]
    const float* be  = (const float*)d_in[3];  // [H]
    const float* Wp  = (const float*)d_in[4];  // [H,P]
    const float* W2  = (const float*)d_in[5];  // [V,H]
    const float* b2  = (const float*)d_in[6];  // [V]
    float* out = (float*)d_out;

    // Workspace layout (4 MB): encP fp32 | predP fp32 | W2 bf16
    float* encP  = (float*)d_ws;                                   // B*T*H
    float* predP = encP + (size_t)Bdim * Tdim * Hdim;              // B*S*H
    unsigned short* W2bf = (unsigned short*)(predP + (size_t)Bdim * Sdim * Hdim); // V*H

    proj_kernel<<<(Bdim * Tdim * Hdim) / 256, 256, 0, stream>>>(
        src, We, be, encP, Bdim * Tdim, Edim, Hdim);
    proj_kernel<<<(Bdim * Sdim * Hdim) / 256, 256, 0, stream>>>(
        tgt, Wp, nullptr, predP, Bdim * Sdim, Pdim, Hdim);
    cvt_bf16_kernel<<<(Vdim * Hdim) / 256, 256, 0, stream>>>(W2, W2bf, Vdim * Hdim);

    // One block per (b, t, 64-s tile): 4*256*2 = 2048 blocks, 16 waves each
    joint_kernel<<<Bdim * Tdim * (Sdim / MTILE), 512, 0, stream>>>(
        encP, predP, W2bf, b2, out);
}
